// FC_KANLayer_69363721830583
// MI455X (gfx1250) — matrix operand order repaired and checked by reading, not compiled, not run on hardware
//
#include <hip/hip_runtime.h>
#include <math.h>

#define D_IN   256
#define D_OUT  512
#define T_LEN  1024
#define NB     6
#define NG     8
#define KSPL   (D_IN * NG)        // 2048
#define GRID_MIN_F (-1.5f)
#define INV_DENOM_F (7.0f / 3.0f)
#define H_F        0.6f           // (GRID_MAX-GRID_MIN)/GRID_SIZE

#define BM 64
#define BN 128
#define KT 32
#define LDP (KT + 4)              // LDS row pad (halves); pairs stay 4B-aligned

typedef __attribute__((ext_vector_type(16))) _Float16 v16h;
typedef __attribute__((ext_vector_type(8)))  float    v8f;

union H2 { _Float16 h[2]; unsigned int u; };

// ---------------------------------------------------------------- LayerNorm
__global__ __launch_bounds__(256) void ln_kernel(
    const float* __restrict__ X, const float* __restrict__ w,
    const float* __restrict__ b, float* __restrict__ Xn)
{
    const int row = blockIdx.x;              // b*T + t
    const int tid = threadIdx.x;             // 256 threads == D_IN
    float v = X[(size_t)row * D_IN + tid];
    float s = v, q = v * v;
    #pragma unroll
    for (int off = 16; off; off >>= 1) {
        s += __shfl_xor(s, off, 32);
        q += __shfl_xor(q, off, 32);
    }
    __shared__ float rs[8], rq[8];
    if ((tid & 31) == 0) { rs[tid >> 5] = s; rq[tid >> 5] = q; }
    __syncthreads();
    s = 0.f; q = 0.f;
    #pragma unroll
    for (int i = 0; i < 8; i++) { s += rs[i]; q += rq[i]; }
    const float mu   = s * (1.0f / D_IN);
    const float var  = q * (1.0f / D_IN) - mu * mu;
    const float rstd = rsqrtf(var + 1e-5f);
    Xn[(size_t)row * D_IN + tid] = (v - mu) * rstd * w[tid] + b[tid];
}

// ------------------------------------------------- cubic B-spline (uniform)
__device__ __forceinline__ float bspline3(float x, int g)
{
    // knots t[j] = GRID_MIN + (j-3)*H ; return order-3 basis g (de Boor)
    const float t0 = GRID_MIN_F + (float)(g - 3) * H_F;
    float b0[4];
    #pragma unroll
    for (int j = 0; j < 4; j++) {
        const float tj = t0 + (float)j * H_F;
        b0[j] = (x >= tj && x < tj + H_F) ? 1.0f : 0.0f;
    }
    const float i1 = 1.0f / H_F, i2 = 1.0f / (2.0f * H_F), i3 = 1.0f / (3.0f * H_F);
    float b1[3];
    #pragma unroll
    for (int j = 0; j < 3; j++)
        b1[j] = ((x - (t0 + j * H_F)) * b0[j] + ((t0 + (j + 2) * H_F) - x) * b0[j + 1]) * i1;
    float b2[2];
    #pragma unroll
    for (int j = 0; j < 2; j++)
        b2[j] = ((x - (t0 + j * H_F)) * b1[j] + ((t0 + (j + 3) * H_F) - x) * b1[j + 1]) * i2;
    return ((x - t0) * b2[0] + ((t0 + 4.0f * H_F) - x) * b2[1]) * i3;
}

// ------------------------------------------------ fused basis + WMMA GEMM
// FUNC: 0 = rbf, 1 = b-spline, 2 = silu(base)
// C[m][n] = sum_k basis(Xn)[m][k] * W[n][k]
template<int FUNC, int KTOT>
__global__ __launch_bounds__(256) void kan_gemm(
    const float* __restrict__ Xn,       // [T_LEN, D_IN]   (one batch)
    const float* __restrict__ W,        // [D_OUT, KTOT]
    const float* __restrict__ grid_rbf, // [NG]
    float* __restrict__ out)            // [T_LEN, D_OUT]  (one batch)
{
    __shared__ _Float16 As[BM][LDP];
    __shared__ _Float16 Bs[BN][LDP];

    const int tid  = threadIdx.x;
    const int lane = tid & 31;
    const int wave = tid >> 5;
    const int wm   = wave & 3;            // 16-row sub-tile
    const int wn   = wave >> 2;           // 64-col half
    const int m0   = blockIdx.y * BM;
    const int n0   = blockIdx.x * BN;

    // A fill: each thread owns one (k, k+1) pair for 4 rows -> packed b32 LDS stores
    const int kp    = tid & 15;           // pair index; k = 2*kp (k0-invariant)
    const int mBase = tid >> 4;           // 0..15
    const int g0    = (2 * kp) & 7;       // even; g1 = g0+1, same d-group always
    const float grf0 = (FUNC == 0) ? grid_rbf[g0]     : 0.0f;
    const float grf1 = (FUNC == 0) ? grid_rbf[g0 + 1] : 0.0f;

    v8f acc[4] = {};

    for (int k0 = 0; k0 < KTOT; k0 += KT) {
        // ---- stage A tile (basis on the fly, packed f16x2 stores)
        #pragma unroll
        for (int i = 0; i < BM / 16; i++) {
            const int m = mBase + i * 16;
            const int k = k0 + 2 * kp;
            float a0, a1;
            if (FUNC == 2) {
                const float2 x2 = *(const float2*)&Xn[(size_t)(m0 + m) * D_IN + k];
                a0 = x2.x / (1.0f + __expf(-x2.x));           // silu
                a1 = x2.y / (1.0f + __expf(-x2.y));
            } else {
                const int d   = k >> 3;                       // shared by the pair
                const float x = Xn[(size_t)(m0 + m) * D_IN + d];
                if (FUNC == 0) {
                    const float u0 = (x - grf0) * INV_DENOM_F;
                    const float u1 = (x - grf1) * INV_DENOM_F;
                    a0 = __expf(-u0 * u0);                    // rbf
                    a1 = __expf(-u1 * u1);
                } else {
                    a0 = bspline3(x, g0);                     // b-spline
                    a1 = bspline3(x, g0 + 1);
                }
            }
            H2 p; p.h[0] = (_Float16)a0; p.h[1] = (_Float16)a1;
            *reinterpret_cast<unsigned int*>(&As[m][2 * kp]) = p.u;
        }
        // ---- stage B tile: float2 loads, packed f16x2 stores (coalesced)
        #pragma unroll
        for (int i = 0; i < (BN * KT) / (2 * 256); i++) {     // 8 iters
            const int e   = tid + i * 256;                    // 0..2047
            const int bkp = e & 15;
            const int n   = e >> 4;
            const float2 w2 =
                *(const float2*)&W[(size_t)(n0 + n) * KTOT + k0 + 2 * bkp];
            H2 p; p.h[0] = (_Float16)w2.x; p.h[1] = (_Float16)w2.y;
            *reinterpret_cast<unsigned int*>(&Bs[n][2 * bkp]) = p.u;
        }
        // ---- prefetch next K-tile of W into cache (global_prefetch_b8)
        if (k0 + KT < KTOT) {
            const int n = tid >> 1;                           // 0..127
            __builtin_prefetch(
                &W[(size_t)(n0 + n) * KTOT + k0 + KT + (tid & 1) * 16], 0, 3);
        }
        __syncthreads();

        // ---- A fragment: CDNA5 16-bit 16x32 layout
        v16h afrag;
        {
            const int m     = wm * 16 + (lane & 15);
            const int khalf = (lane >> 4) * 8;
            #pragma unroll
            for (int i = 0; i < 8; i++) {
                const int k = ((i < 4) ? 2 * i : 16 + 2 * (i - 4)) + khalf;
                afrag[2 * i]     = As[m][k];
                afrag[2 * i + 1] = As[m][k + 1];
            }
        }
        // ---- 4 N-subtiles share the A fragment
        #pragma unroll
        for (int nt = 0; nt < 4; nt++) {
            v16h bfrag;
            const int n  = wn * 64 + nt * 16 + (lane & 15);
            const int kb = (lane >> 4) * 8;
            #pragma unroll
            for (int i = 0; i < 8; i++) {
                const int k = ((i < 4) ? 2 * i : 16 + 2 * (i - 4)) + kb;
                bfrag[2 * i]     = Bs[n][k];
                bfrag[2 * i + 1] = Bs[n][k + 1];
            }
            acc[nt] = __builtin_amdgcn_wmma_f32_16x16x32_f16(
                false, afrag, false, bfrag, (short)0, acc[nt], false, false);
        }
        __syncthreads();
    }

    // ---- store: C/D layout (lane -> N, VGPR r -> M row, half-wave -> +8)
    #pragma unroll
    for (int nt = 0; nt < 4; nt++) {
        const int n = n0 + wn * 64 + nt * 16 + (lane & 15);
        #pragma unroll
        for (int r = 0; r < 8; r++) {
            const int m = m0 + wm * 16 + r + 8 * (lane >> 4);
            out[(size_t)m * D_OUT + n] = acc[nt][r];
        }
    }
}

// ------------------------------------------------------------- DoG wavelet
__global__ __launch_bounds__(256) void dog_kernel(
    const float* __restrict__ Xn2,  // [T_LEN, D_IN] (batch 2)
    const float* __restrict__ bw,   // [D_OUT, D_IN]
    const float* __restrict__ sc,   // [D_OUT, D_IN]
    const float* __restrict__ tr,   // [D_OUT, D_IN]
    float* __restrict__ out2)       // [T_LEN, D_OUT]
{
    __shared__ float xr[D_IN];
    const int t   = blockIdx.x;
    const int tid = threadIdx.x;
    xr[tid] = Xn2[(size_t)t * D_IN + tid];
    __syncthreads();
    #pragma unroll
    for (int oo = 0; oo < 2; oo++) {
        const int o = tid + oo * 256;
        const float* __restrict__ trp = tr + (size_t)o * D_IN;
        const float* __restrict__ scp = sc + (size_t)o * D_IN;
        const float* __restrict__ bwp = bw + (size_t)o * D_IN;
        float acc = 0.0f;
        for (int d = 0; d < D_IN; d++) {
            const float u = (xr[d] - trp[d]) / scp[d];
            acc += (-u) * __expf(-0.5f * u * u) * bwp[d];
        }
        out2[(size_t)t * D_OUT + o] = acc;
    }
}

// ------------------------------------------------------------------ launch
extern "C" void kernel_launch(void* const* d_in, const int* in_sizes, int n_in,
                              void* d_out, int out_size, void* d_ws, size_t ws_size,
                              hipStream_t stream)
{
    const float* X    = (const float*)d_in[0];
    const float* ln_w = (const float*)d_in[1];
    const float* ln_b = (const float*)d_in[2];
    const float* bw   = (const float*)d_in[3];  // base_weight [512,256]
    const float* sw   = (const float*)d_in[4];  // spline_weight [512,2048]
    const float* sc   = (const float*)d_in[5];  // scale [512,256]
    const float* tr   = (const float*)d_in[6];  // translation [512,256]
    const float* grf  = (const float*)d_in[7];  // grid_rbf [8]
    // d_in[8] = grid_bs: uniform knots, reconstructed analytically

    float* out = (float*)d_out;                 // [6,1024,512]
    float* Xn  = (float*)d_ws;                  // [6,1024,256] f32 = 6 MB

    // LayerNorm all 6 batches
    ln_kernel<<<NB * T_LEN, 256, 0, stream>>>(X, ln_w, ln_b, Xn);

    const dim3 gg(D_OUT / BN, T_LEN / BM);      // (4, 16)
    const size_t xs = (size_t)T_LEN * D_IN;     // per-batch Xn stride
    const size_t os = (size_t)T_LEN * D_OUT;    // per-batch out stride

    // FUNC_LIST = ["rbf", "bs", "dog", "base", "rbf", "bs"]
    kan_gemm<0, KSPL><<<gg, 256, 0, stream>>>(Xn + 0 * xs, sw, grf, out + 0 * os);
    kan_gemm<1, KSPL><<<gg, 256, 0, stream>>>(Xn + 1 * xs, sw, grf, out + 1 * os);
    dog_kernel<<<T_LEN, 256, 0, stream>>>(Xn + 2 * xs, bw, sc, tr, out + 2 * os);
    kan_gemm<2, D_IN><<<gg, 256, 0, stream>>>(Xn + 3 * xs, bw, grf, out + 3 * os);
    kan_gemm<0, KSPL><<<gg, 256, 0, stream>>>(Xn + 4 * xs, sw, grf, out + 4 * os);
    kan_gemm<1, KSPL><<<gg, 256, 0, stream>>>(Xn + 5 * xs, sw, grf, out + 5 * os);
}
